// RSSMRollout_81046032875839
// MI455X (gfx1250) — compile-verified, matrix operand order checked by reading, not run on hardware
//
#include <hip/hip_runtime.h>
#include <hip/hip_bf16.h>
#include <stdint.h>

// ---------------- problem constants (match reference) ----------------
#define T_STEPS 50
#define BATCH   1024
#define S_DIM   32
#define D_DIM   600
#define H_DIM   600
#define A_DIM   6
#define E_DIM   1024
#define X_DIM   (A_DIM + S_DIM)   // 38
#define G_DIM   (3 * D_DIM)       // 1800
#define MIN_STD 0.1f

// ---------------- WMMA types ----------------
typedef __attribute__((ext_vector_type(16))) __bf16 v16bf;
typedef __attribute__((ext_vector_type(8)))  float  v8f;

__device__ __forceinline__ int imin(int a, int b) { return a < b ? a : b; }

__device__ __forceinline__ __bf16 f2bf(float f) {          // RNE, bit-exact & libcall-free
  union { float f; uint32_t u; } a; a.f = f;
  uint32_t r = (a.u + 0x7FFFu + ((a.u >> 16) & 1u)) >> 16;
  union { uint16_t s; __bf16 b; } o; o.s = (uint16_t)r;
  return o.b;
}
__device__ __forceinline__ uint32_t pack_bf16x2(float lo, float hi) {
  union { float f; uint32_t u; } a, b; a.f = lo; b.f = hi;
  uint32_t ra = (a.u + 0x7FFFu + ((a.u >> 16) & 1u)) >> 16;
  uint32_t rb = (b.u + 0x7FFFu + ((b.u >> 16) & 1u)) & 0xFFFF0000u;
  return (ra & 0xFFFFu) | rb;
}

__device__ __forceinline__ float dev_elu(float x)      { return x > 0.f ? x : expf(x) - 1.f; }
__device__ __forceinline__ float dev_sigmoid(float x)  { return 1.f / (1.f + expf(-x)); }
__device__ __forceinline__ float dev_softplus(float x) { return fmaxf(x, 0.f) + log1pf(expf(-fabsf(x))); }

// ---------------- CDNA5 async memory->LDS copy (ASYNCcnt path) ----------------
// GLOBAL_LOAD_ASYNC_TO_LDS_B128: per-lane 16-B copy, no VGPR data, tracked by ASYNCcnt.
// Flat LDS pointers truncate to the 32-bit LDS window (ISA 10.2).
__device__ __forceinline__ void async_copy_b128(void* lds_dst, const void* g_src) {
  uint32_t l = (uint32_t)(uintptr_t)lds_dst;
  asm volatile("global_load_async_to_lds_b128 %0, %1, off"
               :: "v"(l), "v"(g_src) : "memory");
}
__device__ __forceinline__ void wait_async_zero() {
  asm volatile("s_wait_asynccnt 0x0" ::: "memory");
}

// ---------------- fp32 -> bf16 conversion (packed stores) ----------------
__global__ void k_f32_to_bf16(const float* __restrict__ src, uint32_t* __restrict__ dst, int npairs, int n) {
  int i = blockIdx.x * blockDim.x + threadIdx.x;   // one thread = 2 elements
  if (i >= npairs) return;
  float lo = src[2 * i];
  float hi = (2 * i + 1 < n) ? src[2 * i + 1] : 0.f;
  dst[i] = pack_bf16x2(lo, hi);
}

// ---------------- assemble x = concat(act_t, stoch) as bf16 ----------------
__global__ void k_assemble_x(const float* __restrict__ act_t, const float* __restrict__ stoch,
                             __bf16* __restrict__ xbuf) {
  int i = blockIdx.x * blockDim.x + threadIdx.x;
  if (i >= BATCH * X_DIM) return;
  int b = i / X_DIM, k = i % X_DIM;
  float a = act_t[b * A_DIM + imin(k, A_DIM - 1)];
  float s = stoch[b * S_DIM + (k >= A_DIM ? k - A_DIM : 0)];
  xbuf[i] = f2bf((k < A_DIM) ? a : s);
}

// ---------------- generic bf16 WMMA GEMM ----------------
// C[MxN] = act( A[MxK](bf16) * W[N x ldw](bf16)^T + bias (+Cin fp32) ); out fp32 or bf16.
#define BM 64
#define BN 128
#define BK 64
#define LDP 72   // LDS row pitch in halves (144 B); fragment sub-offsets stay 16-B aligned

union Stage2 { uint4 q[2]; __bf16 h[16]; };   // A tail staging
union Stage4 { uint4 q[4]; __bf16 h[32]; };   // W tail staging
union Frag   { v16bf v; uint4 q[2]; };

template<int ACT, bool BETA, bool OUTBF>
__global__ __launch_bounds__(256) void k_gemm_bf16(
    const __bf16* __restrict__ A, int lda,
    const __bf16* __restrict__ W, int ldw,
    const float* __restrict__ bias,
    const float* __restrict__ Cin,   // fp32 accumulator input (BETA only)
    void* __restrict__ Cv, int ldc,
    int M, int N, int K)
{
  __shared__ __align__(16) __bf16 As[2][BM * LDP];
  __shared__ __align__(16) __bf16 Bs[2][BN * LDP];

  const int tid = threadIdx.x;
  const int bm0 = blockIdx.y * BM;
  const int bn0 = blockIdx.x * BN;

  // A loader: row = tid/4 (0..63); 16 consecutive halves at col (tid%4)*16
  const int ar  = tid >> 2;
  const int ac  = (tid & 3) * 16;
  const int agm = imin(bm0 + ar, M - 1);          // clamped: OOB rows feed discarded outputs
  // W loader: n = tid/2 (0..127); 32 consecutive halves at col (tid%2)*32
  const int wrn = tid >> 1;
  const int wrc = (tid & 1) * 32;
  const int wgn = imin(bn0 + wrn, N - 1);

  const bool aVec = ((lda & 7) == 0) && ((((uintptr_t)A) & 15) == 0);
  const bool wVec = ((ldw & 7) == 0) && ((((uintptr_t)W) & 15) == 0);

  // wave tiling: 8 waves as 2(M) x 4(N); each wave owns a 32x32 patch = 2x2 WMMA tiles
  const int wid  = tid >> 5;
  const int lane = tid & 31;
  const int wrow = (wid >> 2) * 32;   // 0 / 32
  const int wcol = (wid & 3)  * 32;   // 0 / 32 / 64 / 96
  const int idx  = lane & 15;
  const int half = lane >> 4;

  const v8f vzero = {0.f,0.f,0.f,0.f,0.f,0.f,0.f,0.f};
  v8f acc00 = vzero, acc01 = vzero, acc10 = vzero, acc11 = vzero;

  const int niter = (K + BK - 1) / BK;

  // ---- stage tile k0 into buffer buf ----
  // Fast path: async global->LDS b128 copies (no VGPR data, overlaps the WMMAs).
  // Tail/unaligned path: clamped register loads + b128 LDS stores (rare).
  auto stage = [&](int buf, int k0) {
    if (aVec && (k0 + BK <= K)) {
      const __bf16* g = A + (size_t)agm * lda + k0 + ac;
      __bf16* l = &As[buf][ar * LDP + ac];
      async_copy_b128(l,     g);
      async_copy_b128(l + 8, g + 8);
    } else {
      const __bf16 bz = f2bf(0.0f);
      Stage2 ast;
      #pragma unroll
      for (int j = 0; j < 16; ++j) {
        int gk = k0 + ac + j;
        __bf16 v = A[(size_t)agm * lda + imin(gk, K - 1)];  // clamped, no exec branch
        ast.h[j] = (gk < K) ? v : bz;                       // v_cndmask
      }
      uint4* da = (uint4*)&As[buf][ar * LDP + ac];
      da[0] = ast.q[0]; da[1] = ast.q[1];
    }
    if (wVec && (k0 + BK <= K)) {
      const __bf16* g = W + (size_t)wgn * ldw + k0 + wrc;
      __bf16* l = &Bs[buf][wrn * LDP + wrc];
      #pragma unroll
      for (int j = 0; j < 4; ++j) async_copy_b128(l + 8 * j, g + 8 * j);
    } else {
      const __bf16 bz = f2bf(0.0f);
      Stage4 wst;
      #pragma unroll
      for (int j = 0; j < 32; ++j) {
        int gk = k0 + wrc + j;
        __bf16 v = W[(size_t)wgn * ldw + imin(gk, K - 1)];
        wst.h[j] = (gk < K) ? v : bz;
      }
      uint4* dw = (uint4*)&Bs[buf][wrn * LDP + wrc];
      #pragma unroll
      for (int j = 0; j < 4; ++j) dw[j] = wst.q[j];
    }
  };

  // ---- prologue: stage tile 0 ----
  stage(0, 0);
  wait_async_zero();
  __syncthreads();

  int buf = 0;
  for (int it = 0; it < niter; ++it) {
    const bool has_next = (it + 1 < niter);
    if (has_next) {                      // async copies run in background during the WMMAs
      stage(buf ^ 1, (it + 1) * BK);
      if (it + 2 < niter) {              // keep L2 -> WGP$ ahead of the pipeline
        __builtin_prefetch(A + (size_t)agm * lda + (it + 2) * BK + ac, 0, 1);
        __builtin_prefetch(W + (size_t)wgn * ldw + (it + 2) * BK + wrc, 0, 1);
      }
    }

    // ---- compute: two 32-wide K sub-steps per staged tile ----
    const __bf16* Ab = &As[buf][0];
    const __bf16* Bb = &Bs[buf][0];
    #pragma unroll
    for (int kk = 0; kk < 2; ++kk) {
      const int ko = kk * 32;
      Frag fa0, fa1, fb0, fb1;
      // ISA 16-bit layout: lanes 0-15 hold K {0-7,16-23}, lanes 16-31 hold K {8-15,24-31}
      const __bf16* pa0 = Ab + (wrow + idx)      * LDP + ko;
      const __bf16* pa1 = Ab + (wrow + 16 + idx) * LDP + ko;
      fa0.q[0] = *(const uint4*)(pa0 + half * 8);
      fa0.q[1] = *(const uint4*)(pa0 + 16 + half * 8);
      fa1.q[0] = *(const uint4*)(pa1 + half * 8);
      fa1.q[1] = *(const uint4*)(pa1 + 16 + half * 8);
      const __bf16* pb0 = Bb + (wcol + idx)      * LDP + ko;
      const __bf16* pb1 = Bb + (wcol + 16 + idx) * LDP + ko;
      fb0.q[0] = *(const uint4*)(pb0 + half * 8);
      fb0.q[1] = *(const uint4*)(pb0 + 16 + half * 8);
      fb1.q[0] = *(const uint4*)(pb1 + half * 8);
      fb1.q[1] = *(const uint4*)(pb1 + 16 + half * 8);

      acc00 = __builtin_amdgcn_wmma_f32_16x16x32_bf16(false, fa0.v, false, fb0.v, (short)0, acc00, false, false);
      acc01 = __builtin_amdgcn_wmma_f32_16x16x32_bf16(false, fa0.v, false, fb1.v, (short)0, acc01, false, false);
      acc10 = __builtin_amdgcn_wmma_f32_16x16x32_bf16(false, fa1.v, false, fb0.v, (short)0, acc10, false, false);
      acc11 = __builtin_amdgcn_wmma_f32_16x16x32_bf16(false, fa1.v, false, fb1.v, (short)0, acc11, false, false);
    }

    // each wave drains its own async copies, then block-wide visibility barrier
    wait_async_zero();
    __syncthreads();
    buf ^= 1;
  }

  // ---- epilogue: C/D layout -> lane n = idx; VGPR r -> m = r + 8*half ----
  #pragma unroll
  for (int ti = 0; ti < 2; ++ti) {
    #pragma unroll
    for (int tj = 0; tj < 2; ++tj) {
      v8f a = (ti == 0) ? (tj == 0 ? acc00 : acc01) : (tj == 0 ? acc10 : acc11);
      int gn = bn0 + wcol + tj * 16 + idx;
      bool nok = gn < N;
      float bv = bias ? bias[imin(gn, N - 1)] : 0.0f;
      #pragma unroll
      for (int r = 0; r < 8; ++r) {
        int gm = bm0 + wrow + ti * 16 + r + 8 * half;
        if (nok && gm < M) {
          float v = a[r];
          if (BETA) v += Cin[(size_t)gm * ldc + gn];
          v += bv;
          if (ACT == 1) v = dev_elu(v);
          if (OUTBF) ((__bf16*)Cv)[(size_t)gm * ldc + gn] = f2bf(v);
          else       ((float*)Cv)[(size_t)gm * ldc + gn] = v;
        }
      }
    }
  }
}

// ---------------- GRU gate combine (writes fp32 carry + bf16 GEMM operand) ----------------
__global__ void k_gru(const float* __restrict__ gi, const float* __restrict__ gh,
                      float* __restrict__ det_f32, __bf16* __restrict__ det_bf,
                      float* __restrict__ out_det_a, float* __restrict__ out_det_b) {
  int i = blockIdx.x * blockDim.x + threadIdx.x;
  if (i >= BATCH * D_DIM) return;
  int b = i / D_DIM, d = i % D_DIM;
  size_t base = (size_t)b * G_DIM;
  float r = dev_sigmoid(gi[base + d]            + gh[base + d]);
  float z = dev_sigmoid(gi[base + D_DIM + d]    + gh[base + D_DIM + d]);
  float n = tanhf     (gi[base + 2*D_DIM + d] + r * gh[base + 2*D_DIM + d]);
  float h = det_f32[i];
  float hn = (1.f - z) * n + z * h;
  det_f32[i] = hn;
  det_bf[i]  = f2bf(hn);
  out_det_a[i] = hn;
  out_det_b[i] = hn;
}

// ---------------- distribution heads (prior / posterior) ----------------
__global__ void k_head(const float* __restrict__ p2, const float* __restrict__ noise,
                       float* __restrict__ out_m, float* __restrict__ out_s,
                       float* __restrict__ out_sample, float* __restrict__ stoch_or_null) {
  int i = blockIdx.x * blockDim.x + threadIdx.x;
  if (i >= BATCH * S_DIM) return;
  int b = i / S_DIM, s = i % S_DIM;
  float m   = p2[(size_t)b * 2 * S_DIM + s];
  float raw = p2[(size_t)b * 2 * S_DIM + S_DIM + s];
  float sd  = dev_softplus(raw) + MIN_STD;
  float smp = m + sd * noise[i];
  out_m[i] = m; out_s[i] = sd; out_sample[i] = smp;
  if (stoch_or_null) stoch_or_null[i] = smp;
}

// ---------------- host driver ----------------
static inline void* carve(char* ws, size_t& off, size_t bytes) {
  void* p = ws + off;
  off = (off + bytes + 255) & ~(size_t)255;
  return p;
}

extern "C" void kernel_launch(void* const* d_in, const int* in_sizes, int n_in,
                              void* d_out, int out_size, void* d_ws, size_t ws_size,
                              hipStream_t stream) {
  (void)in_sizes; (void)n_in; (void)out_size; (void)ws_size;

  const float* encoded    = (const float*)d_in[1];
  const float* actions    = (const float*)d_in[2];
  const float* init_stoch = (const float*)d_in[3];
  const float* init_det   = (const float*)d_in[4];
  const float* noise_p    = (const float*)d_in[5];
  const float* noise_q    = (const float*)d_in[6];
  const float* W_in = (const float*)d_in[7];  const float* b_in = (const float*)d_in[8];
  const float* W_ih = (const float*)d_in[9];  const float* W_hh = (const float*)d_in[10];
  const float* b_ih = (const float*)d_in[11]; const float* b_hh = (const float*)d_in[12];
  const float* Wp1  = (const float*)d_in[13]; const float* bp1  = (const float*)d_in[14];
  const float* Wp2  = (const float*)d_in[15]; const float* bp2  = (const float*)d_in[16];
  const float* Wq1  = (const float*)d_in[17]; const float* bq1  = (const float*)d_in[18];
  const float* Wq2  = (const float*)d_in[19]; const float* bq2  = (const float*)d_in[20];

  // ---- scratch layout ----
  char* ws = (char*)d_ws; size_t off = 0;
  __bf16* wih_bf = (__bf16*)carve(ws, off, (size_t)G_DIM * H_DIM * 2);
  __bf16* whh_bf = (__bf16*)carve(ws, off, (size_t)G_DIM * D_DIM * 2);
  __bf16* wp1_bf = (__bf16*)carve(ws, off, (size_t)H_DIM * D_DIM * 2);
  __bf16* wp2_bf = (__bf16*)carve(ws, off, (size_t)2 * S_DIM * H_DIM * 2);
  __bf16* wq1_bf = (__bf16*)carve(ws, off, (size_t)H_DIM * (D_DIM + E_DIM) * 2);
  __bf16* wq2_bf = (__bf16*)carve(ws, off, (size_t)2 * S_DIM * H_DIM * 2);
  __bf16* win_bf = (__bf16*)carve(ws, off, (size_t)H_DIM * X_DIM * 2);
  // bf16 activations (GEMM A operands)
  __bf16* xbuf   = (__bf16*)carve(ws, off, (size_t)BATCH * X_DIM * 2);
  __bf16* rnn_bf = (__bf16*)carve(ws, off, (size_t)BATCH * H_DIM * 2);
  __bf16* det_bf = (__bf16*)carve(ws, off, (size_t)BATCH * D_DIM * 2);
  __bf16* p1_bf  = (__bf16*)carve(ws, off, (size_t)BATCH * H_DIM * 2);
  __bf16* q1_bf  = (__bf16*)carve(ws, off, (size_t)BATCH * H_DIM * 2);
  __bf16* enc_bf = (__bf16*)carve(ws, off, (size_t)BATCH * E_DIM * 2);
  // fp32 buffers
  float* gi     = (float*)carve(ws, off, (size_t)BATCH * G_DIM * 4);
  float* gh     = (float*)carve(ws, off, (size_t)BATCH * G_DIM * 4);
  float* det    = (float*)carve(ws, off, (size_t)BATCH * D_DIM * 4);
  float* stoch  = (float*)carve(ws, off, (size_t)BATCH * S_DIM * 4);
  float* q1_f32 = (float*)carve(ws, off, (size_t)BATCH * H_DIM * 4);   // split-K partial
  float* p2     = (float*)carve(ws, off, (size_t)BATCH * 2 * S_DIM * 4);
  float* q2     = (float*)carve(ws, off, (size_t)BATCH * 2 * S_DIM * 4);

  // ---- output layout (outs concatenated flat in return order) ----
  const size_t TBS = (size_t)T_STEPS * BATCH * S_DIM;
  const size_t TBD = (size_t)T_STEPS * BATCH * D_DIM;
  float* out  = (float*)d_out;
  float* out0 = out;               // pm
  float* out1 = out0 + TBS;        // ps
  float* out2 = out1 + TBS;        // prior_stoch
  float* out3 = out2 + TBS;        // det_new
  float* out4 = out3 + TBD;        // qm
  float* out5 = out4 + TBS;        // qs
  float* out6 = out5 + TBS;        // post_stoch
  float* out7 = out6 + TBS;        // det_new (again)

  auto conv = [&](const float* s, __bf16* d, int n) {
    int npairs = (n + 1) / 2;
    k_f32_to_bf16<<<(npairs + 255) / 256, 256, 0, stream>>>(s, (uint32_t*)d, npairs, n);
  };
  // one-time weight conversions (idempotent, graph-safe)
  conv(W_ih, wih_bf, G_DIM * H_DIM);
  conv(W_hh, whh_bf, G_DIM * D_DIM);
  conv(Wp1,  wp1_bf, H_DIM * D_DIM);
  conv(Wp2,  wp2_bf, 2 * S_DIM * H_DIM);
  conv(Wq1,  wq1_bf, H_DIM * (D_DIM + E_DIM));
  conv(Wq2,  wq2_bf, 2 * S_DIM * H_DIM);
  conv(W_in, win_bf, H_DIM * X_DIM);
  conv(init_det, det_bf, BATCH * D_DIM);

  hipMemcpyAsync(stoch, init_stoch, (size_t)BATCH * S_DIM * 4, hipMemcpyDeviceToDevice, stream);
  hipMemcpyAsync(det,   init_det,   (size_t)BATCH * D_DIM * 4, hipMemcpyDeviceToDevice, stream);

  const dim3 blk(256);
  const dim3 gH((H_DIM + BN - 1) / BN, (BATCH + BM - 1) / BM);     // N=600
  const dim3 gG((G_DIM + BN - 1) / BN, (BATCH + BM - 1) / BM);     // N=1800
  const dim3 gS((2 * S_DIM + BN - 1) / BN, (BATCH + BM - 1) / BM); // N=64

  for (int t = 0; t < T_STEPS; ++t) {
    const float* act_t = actions + (size_t)t * BATCH * A_DIM;
    const float* enc_t = encoded + (size_t)t * BATCH * E_DIM;
    const float* np_t  = noise_p + (size_t)t * BATCH * S_DIM;
    const float* nq_t  = noise_q + (size_t)t * BATCH * S_DIM;
    float* o0 = out0 + (size_t)t * BATCH * S_DIM;
    float* o1 = out1 + (size_t)t * BATCH * S_DIM;
    float* o2 = out2 + (size_t)t * BATCH * S_DIM;
    float* o3 = out3 + (size_t)t * BATCH * D_DIM;
    float* o4 = out4 + (size_t)t * BATCH * S_DIM;
    float* o5 = out5 + (size_t)t * BATCH * S_DIM;
    float* o6 = out6 + (size_t)t * BATCH * S_DIM;
    float* o7 = out7 + (size_t)t * BATCH * D_DIM;

    // per-step input conversions
    conv(enc_t, enc_bf, BATCH * E_DIM);
    k_assemble_x<<<(BATCH * X_DIM + 255) / 256, blk, 0, stream>>>(act_t, stoch, xbuf);

    // rnn_in = elu([act, stoch] @ W_in^T + b_in)   -> bf16
    k_gemm_bf16<1, false, true><<<gH, blk, 0, stream>>>(xbuf, X_DIM, win_bf, X_DIM, b_in,
                                                        nullptr, rnn_bf, H_DIM, BATCH, H_DIM, X_DIM);
    // GRU gates (fp32 outputs for the gate math)
    k_gemm_bf16<0, false, false><<<gG, blk, 0, stream>>>(rnn_bf, H_DIM, wih_bf, H_DIM, b_ih,
                                                         nullptr, gi, G_DIM, BATCH, G_DIM, H_DIM);
    k_gemm_bf16<0, false, false><<<gG, blk, 0, stream>>>(det_bf, D_DIM, whh_bf, D_DIM, b_hh,
                                                         nullptr, gh, G_DIM, BATCH, G_DIM, D_DIM);
    k_gru<<<(BATCH * D_DIM + 255) / 256, blk, 0, stream>>>(gi, gh, det, det_bf, o3, o7);

    // prior head
    k_gemm_bf16<1, false, true><<<gH, blk, 0, stream>>>(det_bf, D_DIM, wp1_bf, D_DIM, bp1,
                                                        nullptr, p1_bf, H_DIM, BATCH, H_DIM, D_DIM);
    k_gemm_bf16<0, false, false><<<gS, blk, 0, stream>>>(p1_bf, H_DIM, wp2_bf, H_DIM, bp2,
                                                         nullptr, p2, 2 * S_DIM, BATCH, 2 * S_DIM, H_DIM);
    k_head<<<(BATCH * S_DIM + 255) / 256, blk, 0, stream>>>(p2, np_t, o0, o1, o2, nullptr);

    // posterior head: concat(det, enc) as split-K accumulation (fp32 partial -> bf16 final)
    k_gemm_bf16<0, false, false><<<gH, blk, 0, stream>>>(det_bf, D_DIM, wq1_bf, D_DIM + E_DIM, nullptr,
                                                         nullptr, q1_f32, H_DIM, BATCH, H_DIM, D_DIM);
    k_gemm_bf16<1, true, true><<<gH, blk, 0, stream>>>(enc_bf, E_DIM, wq1_bf + D_DIM, D_DIM + E_DIM, bq1,
                                                       q1_f32, q1_bf, H_DIM, BATCH, H_DIM, E_DIM);
    k_gemm_bf16<0, false, false><<<gS, blk, 0, stream>>>(q1_bf, H_DIM, wq2_bf, H_DIM, bq2,
                                                         nullptr, q2, 2 * S_DIM, BATCH, 2 * S_DIM, H_DIM);
    k_head<<<(BATCH * S_DIM + 255) / 256, blk, 0, stream>>>(q2, nq_t, o4, o5, o6, stoch);
  }
}
